// PCNN_472446403144
// MI455X (gfx1250) — matrix operand order repaired and checked by reading, not compiled, not run on hardware
//
#include <hip/hip_runtime.h>

typedef __attribute__((ext_vector_type(16))) _Float16 v16h;
typedef __attribute__((ext_vector_type(2)))  _Float16 h2;
typedef __attribute__((ext_vector_type(8)))  float    v8f;

// exact match for the async-LDS builtin's parameter type:
// 'int __attribute__((vector_size(16)))' in AS1 (global) / AS3 (LDS)
typedef int v4i_vs __attribute__((vector_size(16)));
typedef __attribute__((address_space(1))) v4i_vs* as1_v4i_ptr;
typedef __attribute__((address_space(3))) v4i_vs* as3_v4i_ptr;

static constexpr int   BATCH   = 8;
static constexpr float SIGMA0  = 0.03125f;                        // 1/sqrt(1024), fixed (faithful)
static constexpr float INV_2S2 = 1.0f / (2.0f * SIGMA0 * SIGMA0); // 512
static constexpr float TSPACE  = 2.0f * SIGMA0;                   // SPACING * sigma

#define HAS_ASYNC_LDS (__has_builtin(__builtin_amdgcn_global_load_async_to_lds_b128) && \
                       __has_builtin(__builtin_amdgcn_s_wait_asynccnt))

// 16-byte global -> LDS async DMA (ASYNCcnt tracked); fallback = regular copy.
__device__ __forceinline__ void async_copy_b128(const _Float16* g, _Float16* lds)
{
#if HAS_ASYNC_LDS
    __builtin_amdgcn_global_load_async_to_lds_b128(
        (as1_v4i_ptr)g, (as3_v4i_ptr)lds, 0, 0);
#else
    *(uint4*)lds = *(const uint4*)g;
#endif
}
__device__ __forceinline__ void async_join()
{
#if HAS_ASYNC_LDS
    __builtin_amdgcn_s_wait_asynccnt(0);
#endif
}

// ---------------------------------------------------------------------------
// Fused PCNN extension-operator conv:
//   out = relu( (1/N) * sum_l (e_l @ feat) @ W_l + bias )
// featT: f16 channel-major features [B][CinP][N] (pad rows zeroed) so stage-A
//        B-fragments load contiguous K-pairs.
// wT16:  f16 output-major weights [27][Cout][Cin] so the stage-B LDS tile is
//        [64][CinP] with contiguous K-pairs per output column.
// Block: 128 thr (4 wave32). Grid: (B * N/16, Cout/64).
// ---------------------------------------------------------------------------
template<int N, int Cin, int CinP, int Cout>
__global__ __launch_bounds__(128)
void pcnn_conv_wmma(const float* __restrict__ pts,        // [B,N,3]
                    const _Float16* __restrict__ featT,   // [B,CinP,N]
                    const _Float16* __restrict__ wT16,    // [27,Cout,Cin]
                    const float* __restrict__ bias,       // [Cout]
                    float* __restrict__ out,              // [B,N,Cout]
                    _Float16* __restrict__ outT)          // [B,Cout,N] or null
{
    constexpr int NT  = N / 16;     // i tiles
    constexpr int NCC = CinP / 16;  // channel chunks
    constexpr int NJ  = N / 32;     // j tiles (K=32 per WMMA)

    const int b     = blockIdx.x / NT;
    const int itile = blockIdx.x % NT;
    const int cg    = blockIdx.y;          // 64-wide cout group
    const int tid   = threadIdx.x;
    const int wave  = tid >> 5;
    const int lane  = tid & 31;
    const int hi    = lane >> 4;           // half-wave selector
    const int m16   = lane & 15;

    __shared__ alignas(16) float    p_lds4[N * 4];     // xyz + pad -> b128 loads
    __shared__ alignas(16) float    s_f32[16 * CinP];
    __shared__ alignas(16) _Float16 s_f16[16 * CinP];
    __shared__ alignas(16) _Float16 w_ldsT[64 * CinP]; // [o][c], K contiguous

    for (int i = tid; i < N; i += 128) {
        p_lds4[i * 4 + 0] = pts[(b * N + i) * 3 + 0];
        p_lds4[i * 4 + 1] = pts[(b * N + i) * 3 + 1];
        p_lds4[i * 4 + 2] = pts[(b * N + i) * 3 + 2];
        p_lds4[i * 4 + 3] = 0.f;
    }
    __syncthreads();

    // Row point for this lane's M index (A layout uses M = lane&15)
    const int   row = itile * 16 + m16;
    const float pix = p_lds4[row * 4 + 0];
    const float piy = p_lds4[row * 4 + 1];
    const float piz = p_lds4[row * 4 + 2];

    const _Float16* ftb  = featT + (size_t)b * CinP * N;
    const _Float16* wrow = &w_ldsT[(wave * 16 + m16) * CinP];   // this lane's o-column

    const v8f zero8 = {0.f, 0.f, 0.f, 0.f, 0.f, 0.f, 0.f, 0.f};
    v8f accO = zero8;   // persistent 16x16 f32 output fragment (cout chunk = wave)

    for (int l = 0; l < 27; ++l) {
        const float tx = TSPACE * (float)((l / 9) - 1);
        const float ty = TSPACE * (float)(((l / 3) % 3) - 1);
        const float tz = TSPACE * (float)((l % 3) - 1);
        const float qx = pix + tx, qy = piy + ty, qz = piz + tz;

        // zero S staging
        for (int i = tid; i < 16 * CinP; i += 128) s_f32[i] = 0.f;

        // weight tile [64][Cin] via async LDS DMA (16B chunks; rows = Cin*2 B).
        if constexpr (Cin >= 8) {
            constexpr int CH = Cin / 8;        // 16B chunks per row
            for (int idx = tid; idx < 64 * CH; idx += 128) {
                const int o  = idx / CH;
                const int c8 = (idx % CH) * 8;
                async_copy_b128(&wT16[((size_t)l * Cout + cg * 64 + o) * Cin + c8],
                                &w_ldsT[o * CinP + c8]);
            }
        } else {  // layer 0: rows of 4 f16 (8B) — scalar copy
            for (int idx = tid; idx < 64 * Cin; idx += 128) {
                const int o = idx / Cin, c = idx % Cin;
                w_ldsT[o * CinP + c] = wT16[((size_t)l * Cout + cg * 64 + o) * Cin + c];
            }
        }
        if constexpr (CinP > Cin) {   // zero-pad K columns (layer 0 only)
            for (int idx = tid; idx < 64 * (CinP - Cin); idx += 128) {
                const int o = idx / (CinP - Cin);
                const int c = Cin + idx % (CinP - Cin);
                w_ldsT[o * CinP + c] = (_Float16)0.f;
            }
        }
        async_join();
        __syncthreads();

        // ---------------- Stage A: S_l = e_l @ feat (waves split j) ----------
        v8f accS[NCC];
        #pragma unroll
        for (int cc = 0; cc < NCC; ++cc) accS[cc] = zero8;

        for (int jt = wave; jt < NJ; jt += 4) {
            const int j0 = jt * 32;
            // A fragment: e values, 16x32 f16 layout
            //   lanes 0-15 : K in {0..7, 16..23};  lanes 16-31: K in {8..15, 24..31}
            v16h af;
            #pragma unroll
            for (int v = 0; v < 8; ++v) {
                const int kb = ((v < 4) ? (2 * v) : (16 + 2 * (v - 4))) + (hi ? 8 : 0);
                #pragma unroll
                for (int h = 0; h < 2; ++h) {
                    const int j = j0 + kb + h;
                    const float4 pj = *(const float4*)&p_lds4[j * 4];  // one ds_load_b128
                    const float dx = qx - pj.x;
                    const float dy = qy - pj.y;
                    const float dz = qz - pj.z;
                    const float d2 = dx * dx + dy * dy + dz * dz;
                    af[2 * v + h] = (_Float16)__expf(-d2 * INV_2S2);
                }
            }
            #pragma unroll
            for (int cc = 0; cc < NCC; ++cc) {
                const int c = cc * 16 + m16;
                const _Float16* fp = ftb + (size_t)c * N + j0;
                // B fragment: feat 32x16 f16 layout; K-pairs contiguous in featT
                v16h bf;
                #pragma unroll
                for (int v = 0; v < 8; ++v) {
                    const int k = ((v < 4) ? (2 * v) : (16 + 2 * (v - 4))) + (hi ? 8 : 0);
                    const h2 p = *(const h2*)(fp + k);
                    bf[2 * v]     = p[0];
                    bf[2 * v + 1] = p[1];
                }
                accS[cc] = __builtin_amdgcn_wmma_f32_16x16x32_f16(
                    false, af, false, bf, (short)0, accS[cc], false, false);
            }
        }

        // reduce partial S across waves (C layout: VGPR r -> row r / r+8)
        #pragma unroll
        for (int cc = 0; cc < NCC; ++cc) {
            const int col = cc * 16 + m16;
            #pragma unroll
            for (int r = 0; r < 8; ++r)
                atomicAdd(&s_f32[(r + (hi ? 8 : 0)) * CinP + col], accS[cc][r]);
        }
        __syncthreads();
        for (int i = tid; i < 16 * CinP; i += 128) s_f16[i] = (_Float16)s_f32[i];
        __syncthreads();

        // ---------------- Stage B: accO += S_l @ W_l (wave = cout chunk) -----
        #pragma unroll
        for (int k0 = 0; k0 < CinP; k0 += 32) {
            v16h a2, b2;
            #pragma unroll
            for (int v = 0; v < 8; ++v) {
                const int kb = ((v < 4) ? (2 * v) : (16 + 2 * (v - 4))) + (hi ? 8 : 0);
                const h2 pa = *(const h2*)&s_f16[m16 * CinP + k0 + kb];
                a2[2 * v]     = pa[0];
                a2[2 * v + 1] = pa[1];
                const int k = ((v < 4) ? (2 * v) : (16 + 2 * (v - 4))) + (hi ? 8 : 0);
                const h2 pb = *(const h2*)&wrow[k0 + k];     // K-pairs contiguous
                b2[2 * v]     = pb[0];
                b2[2 * v + 1] = pb[1];
            }
            accO = __builtin_amdgcn_wmma_f32_16x16x32_f16(
                false, a2, false, b2, (short)0, accO, false, false);
        }
        __syncthreads();   // protect s_f32/w_ldsT before next l
    }

    // epilogue: scale, bias, relu; also emit f16 channel-major copy if chained
    const float invn = 1.0f / (float)N;
    const int   ocol = cg * 64 + wave * 16 + m16;
    const float bv   = bias[ocol];
    #pragma unroll
    for (int r = 0; r < 8; ++r) {
        const int orow = itile * 16 + r + (hi ? 8 : 0);
        float v = accO[r] * invn + bv;
        v = (v > 0.f) ? v : 0.f;
        out[(b * N + orow) * Cout + ocol] = v;
        if (outT) outT[((size_t)b * Cout + ocol) * N + orow] = (_Float16)v;
    }
}

// ---------------------------------------------------------------------------
// Helper kernels
// ---------------------------------------------------------------------------
// w [27,Cin,Cout] f32 -> wT [27,Cout,Cin] f16
__global__ void cvt_w_transpose(const float* __restrict__ src, _Float16* __restrict__ dst,
                                int Cin, int Cout, int total)
{
    int gid = blockIdx.x * blockDim.x + threadIdx.x;
    if (gid >= total) return;
    const int l = gid / (Cout * Cin);
    const int r = gid % (Cout * Cin);
    const int o = r / Cin, c = r % Cin;
    dst[gid] = (_Float16)src[((size_t)l * Cin + c) * Cout + o];
}

__global__ void fill_zero_f32(float* __restrict__ p, int n)
{
    int i = blockIdx.x * blockDim.x + threadIdx.x;
    if (i < n) p[i] = 0.f;
}

// feat [B,Npts,C] f32 -> featT [B,C,Npts] f16  (C == next layer CinP)
__global__ void transpose_to_f16(const float* __restrict__ src, _Float16* __restrict__ dst,
                                 int Npts, int C)
{
    int gid = blockIdx.x * blockDim.x + threadIdx.x;
    if (gid >= BATCH * C * Npts) return;
    const int b = gid / (C * Npts);
    const int r = gid % (C * Npts);
    const int c = r / Npts;
    const int n = r % Npts;
    dst[gid] = (_Float16)src[(b * Npts + n) * C + c];
}

// pc [B,3,1024] -> pts [B,1024,3], featT0 [B,32,1024] = (x,y,z,1, 0-pad)
__global__ void init_pts_featT(const float* __restrict__ pc,
                               float* __restrict__ pts, _Float16* __restrict__ featT)
{
    int i = blockIdx.x * blockDim.x + threadIdx.x;
    if (i >= BATCH * 1024) return;
    const int b = i >> 10, n = i & 1023;
    const float x = pc[(b * 3 + 0) * 1024 + n];
    const float y = pc[(b * 3 + 1) * 1024 + n];
    const float z = pc[(b * 3 + 2) * 1024 + n];
    pts[i * 3 + 0] = x; pts[i * 3 + 1] = y; pts[i * 3 + 2] = z;
    _Float16* ft = featT + (size_t)b * 32 * 1024;
    ft[0 * 1024 + n] = (_Float16)x;
    ft[1 * 1024 + n] = (_Float16)y;
    ft[2 * 1024 + n] = (_Float16)z;
    ft[3 * 1024 + n] = (_Float16)1.f;
    for (int c = 4; c < 32; ++c) ft[c * 1024 + n] = (_Float16)0.f;
}

// Farthest point sampling; one block per batch element.
template<int N>
__global__ __launch_bounds__(256)
void fps_kernel(const float* __restrict__ pts, float* __restrict__ centers, int m)
{
    const int b = blockIdx.x, tid = threadIdx.x;
    __shared__ float px[N], py[N], pz[N], dd[N];
    __shared__ float rv[256];
    __shared__ int   ri[256];

    for (int j = tid; j < N; j += 256) {
        px[j] = pts[(b * N + j) * 3 + 0];
        py[j] = pts[(b * N + j) * 3 + 1];
        pz[j] = pts[(b * N + j) * 3 + 2];
    }
    __syncthreads();
    const float x0 = px[0], y0 = py[0], z0 = pz[0];
    for (int j = tid; j < N; j += 256) {
        const float dx = px[j] - x0, dy = py[j] - y0, dz = pz[j] - z0;
        dd[j] = dx * dx + dy * dy + dz * dz;
    }
    if (tid == 0) {
        centers[(b * m + 0) * 3 + 0] = x0;
        centers[(b * m + 0) * 3 + 1] = y0;
        centers[(b * m + 0) * 3 + 2] = z0;
    }
    __syncthreads();

    for (int it = 1; it < m; ++it) {
        float bv = -1.f; int bi = 0;
        for (int j = tid; j < N; j += 256) {
            const float v = dd[j];
            if (v > bv) { bv = v; bi = j; }
        }
        rv[tid] = bv; ri[tid] = bi;
        __syncthreads();
        for (int s = 128; s > 0; s >>= 1) {
            if (tid < s) {
                const float v2 = rv[tid + s]; const int i2 = ri[tid + s];
                if (v2 > rv[tid] || (v2 == rv[tid] && i2 < ri[tid])) { rv[tid] = v2; ri[tid] = i2; }
            }
            __syncthreads();
        }
        const int nxt = ri[0];
        const float cx = px[nxt], cy = py[nxt], cz = pz[nxt];
        for (int j = tid; j < N; j += 256) {
            const float dx = px[j] - cx, dy = py[j] - cy, dz = pz[j] - cz;
            const float nd = dx * dx + dy * dy + dz * dz;
            if (nd < dd[j]) dd[j] = nd;
        }
        if (tid == 0) {
            centers[(b * m + it) * 3 + 0] = cx;
            centers[(b * m + it) * 3 + 1] = cy;
            centers[(b * m + it) * 3 + 2] = cz;
        }
        __syncthreads();
    }
}

// Nearest-center assignment + segment max-pool. Features are post-ReLU (>= 0),
// so uint atomicMax on the IEEE bit pattern == float max; pooled pre-zeroed.
template<int N, int C>
__global__ __launch_bounds__(256)
void pool_max_kernel(const float* __restrict__ pts, const float* __restrict__ centers,
                     const float* __restrict__ feat, float* __restrict__ pooled, int m)
{
    const int b = blockIdx.x, tid = threadIdx.x;
    __shared__ float cx[256], cy[256], cz[256];
    for (int i = tid; i < m; i += 256) {
        cx[i] = centers[(b * m + i) * 3 + 0];
        cy[i] = centers[(b * m + i) * 3 + 1];
        cz[i] = centers[(b * m + i) * 3 + 2];
    }
    __syncthreads();
    for (int j = tid; j < N; j += 256) {
        const float x = pts[(b * N + j) * 3 + 0];
        const float y = pts[(b * N + j) * 3 + 1];
        const float z = pts[(b * N + j) * 3 + 2];
        int best = 0; float bd = 3.4e38f;
        for (int i = 0; i < m; ++i) {
            const float dx = x - cx[i], dy = y - cy[i], dz = z - cz[i];
            const float d = dx * dx + dy * dy + dz * dz;
            if (d < bd) { bd = d; best = i; }
        }
        const float* fr = &feat[(b * N + j) * C];
        unsigned* pr = (unsigned*)&pooled[(b * m + best) * C];
        for (int c = 0; c < C; ++c)
            atomicMax(&pr[c], __float_as_uint(fr[c]));
    }
}

// y[8,M] = x[8,K] @ w[K,M] (+ optional bias)
__global__ void fc_kernel(const float* __restrict__ x, const float* __restrict__ w,
                          const float* __restrict__ bias, float* __restrict__ y,
                          int K, int M)
{
    const int gid = blockIdx.x * blockDim.x + threadIdx.x;
    if (gid >= BATCH * M) return;
    const int r = gid / M, c = gid % M;
    float acc = bias ? bias[c] : 0.f;
    for (int k = 0; k < K; ++k) acc += x[r * K + k] * w[k * M + c];
    y[gid] = acc;
}

// In-place batch-norm (over batch of 8) + ReLU, per column.
__global__ void bn_relu_kernel(float* __restrict__ y, const float* __restrict__ g,
                               const float* __restrict__ be, int M)
{
    const int c = blockIdx.x * blockDim.x + threadIdx.x;
    if (c >= M) return;
    float mu = 0.f;
    for (int r = 0; r < BATCH; ++r) mu += y[r * M + c];
    mu *= (1.0f / BATCH);
    float var = 0.f;
    for (int r = 0; r < BATCH; ++r) { const float t = y[r * M + c] - mu; var += t * t; }
    var *= (1.0f / BATCH);
    const float s = rsqrtf(var + 1e-5f);
    for (int r = 0; r < BATCH; ++r) {
        const float t = g[c] * (y[r * M + c] - mu) * s + be[c];
        y[r * M + c] = (t > 0.f) ? t : 0.f;
    }
}

// ---------------------------------------------------------------------------
extern "C" void kernel_launch(void* const* d_in, const int* in_sizes, int n_in,
                              void* d_out, int out_size, void* d_ws, size_t ws_size,
                              hipStream_t stream)
{
    (void)in_sizes; (void)n_in; (void)out_size; (void)ws_size;

    const float* pc = (const float*)d_in[0];
    const float* w_c[6] = {(const float*)d_in[1], (const float*)d_in[3], (const float*)d_in[5],
                           (const float*)d_in[7], (const float*)d_in[9], (const float*)d_in[11]};
    const float* b_c[6] = {(const float*)d_in[2], (const float*)d_in[4], (const float*)d_in[6],
                           (const float*)d_in[8], (const float*)d_in[10], (const float*)d_in[12]};
    const float* w_fc1 = (const float*)d_in[13];
    const float* g_fc1 = (const float*)d_in[14];
    const float* be_fc1 = (const float*)d_in[15];
    const float* w_fc2 = (const float*)d_in[16];
    const float* g_fc2 = (const float*)d_in[17];
    const float* be_fc2 = (const float*)d_in[18];
    const float* w_fc3 = (const float*)d_in[19];
    const float* b_fc3 = (const float*)d_in[20];

    // ---- workspace carve (bump allocator, 256B aligned) ----
    size_t cur = 0;
    auto alloc = [&](size_t bytes) -> void* {
        void* p = (char*)d_ws + cur;
        cur = (cur + bytes + 255) & ~(size_t)255;
        return p;
    };
    float* pts0   = (float*)alloc((size_t)BATCH * 1024 * 3 * 4);
    float* ctr256 = (float*)alloc((size_t)BATCH * 256 * 3 * 4);
    float* ctr64  = (float*)alloc((size_t)BATCH * 64 * 3 * 4);
    float* ctr1   = (float*)alloc((size_t)BATCH * 1 * 3 * 4);
    float* featA  = (float*)alloc((size_t)BATCH * 1024 * 64 * 4);     // 2 MB f32 ping
    float* featB  = (float*)alloc((size_t)BATCH * 1024 * 64 * 4);     // 2 MB f32 pong
    _Float16* featT_A = (_Float16*)alloc((size_t)BATCH * 64 * 1024 * 2);  // 1 MB f16 ping
    _Float16* featT_B = (_Float16*)alloc((size_t)BATCH * 64 * 1024 * 2);  // 1 MB f16 pong
    static const int wcnt[6] = {27 * 4 * 64, 27 * 64 * 64, 27 * 64 * 128,
                                27 * 128 * 128, 27 * 128 * 256, 27 * 256 * 1024};
    static const int wcin[6]  = {4, 64, 64, 128, 128, 256};
    static const int wcout[6] = {64, 64, 128, 128, 256, 1024};
    _Float16* w16[6];
    for (int i = 0; i < 6; ++i) w16[i] = (_Float16*)alloc((size_t)wcnt[i] * 2);
    float* fcbuf1 = (float*)alloc((size_t)BATCH * 512 * 4);
    float* fcbuf2 = (float*)alloc((size_t)BATCH * 256 * 4);

    // ---- convert conv weights to f16, output-major [27,Cout,Cin] ----
    for (int i = 0; i < 6; ++i)
        cvt_w_transpose<<<(wcnt[i] + 255) / 256, 256, 0, stream>>>(
            w_c[i], w16[i], wcin[i], wcout[i], wcnt[i]);

    // ---- input transpose + padded channel-major f16 features ----
    init_pts_featT<<<(BATCH * 1024 + 255) / 256, 256, 0, stream>>>(pc, pts0, featT_A);

    // ---- block 0 @ N=1024 ----
    pcnn_conv_wmma<1024, 4, 32, 64><<<dim3(BATCH * 64, 1), 128, 0, stream>>>(
        pts0, featT_A, w16[0], b_c[0], featB, featT_B);
    pcnn_conv_wmma<1024, 64, 64, 64><<<dim3(BATCH * 64, 1), 128, 0, stream>>>(
        pts0, featT_B, w16[1], b_c[1], featA, nullptr);

    fps_kernel<1024><<<BATCH, 256, 0, stream>>>(pts0, ctr256, 256);
    fill_zero_f32<<<(BATCH * 256 * 64 + 255) / 256, 256, 0, stream>>>(featB, BATCH * 256 * 64);
    pool_max_kernel<1024, 64><<<BATCH, 256, 0, stream>>>(pts0, ctr256, featA, featB, 256);
    transpose_to_f16<<<(BATCH * 64 * 256 + 255) / 256, 256, 0, stream>>>(featB, featT_A, 256, 64);

    // ---- block 1 @ N=256 ----
    pcnn_conv_wmma<256, 64, 64, 128><<<dim3(BATCH * 16, 2), 128, 0, stream>>>(
        ctr256, featT_A, w16[2], b_c[2], featA, featT_B);
    pcnn_conv_wmma<256, 128, 128, 128><<<dim3(BATCH * 16, 2), 128, 0, stream>>>(
        ctr256, featT_B, w16[3], b_c[3], featB, nullptr);

    fps_kernel<256><<<BATCH, 256, 0, stream>>>(ctr256, ctr64, 64);
    fill_zero_f32<<<(BATCH * 64 * 128 + 255) / 256, 256, 0, stream>>>(featA, BATCH * 64 * 128);
    pool_max_kernel<256, 128><<<BATCH, 256, 0, stream>>>(ctr256, ctr64, featB, featA, 64);
    transpose_to_f16<<<(BATCH * 128 * 64 + 255) / 256, 256, 0, stream>>>(featA, featT_A, 64, 128);

    // ---- block 2 @ N=64 ----
    pcnn_conv_wmma<64, 128, 128, 256><<<dim3(BATCH * 4, 4), 128, 0, stream>>>(
        ctr64, featT_A, w16[4], b_c[4], featB, featT_B);
    pcnn_conv_wmma<64, 256, 256, 1024><<<dim3(BATCH * 4, 16), 128, 0, stream>>>(
        ctr64, featT_B, w16[5], b_c[5], featA, nullptr);

    fps_kernel<64><<<BATCH, 256, 0, stream>>>(ctr64, ctr1, 1);
    fill_zero_f32<<<(BATCH * 1024 + 255) / 256, 256, 0, stream>>>(featB, BATCH * 1024);
    pool_max_kernel<64, 1024><<<BATCH, 256, 0, stream>>>(ctr64, ctr1, featA, featB, 1);

    // ---- FC head: 1024 -> 512 -> 256 -> 40 ----
    fc_kernel<<<(BATCH * 512 + 255) / 256, 256, 0, stream>>>(featB, w_fc1, nullptr, fcbuf1, 1024, 512);
    bn_relu_kernel<<<(512 + 255) / 256, 256, 0, stream>>>(fcbuf1, g_fc1, be_fc1, 512);
    fc_kernel<<<(BATCH * 256 + 255) / 256, 256, 0, stream>>>(fcbuf1, w_fc2, nullptr, fcbuf2, 512, 256);
    bn_relu_kernel<<<(256 + 255) / 256, 256, 0, stream>>>(fcbuf2, g_fc2, be_fc2, 256);
    fc_kernel<<<(BATCH * 40 + 255) / 256, 256, 0, stream>>>(fcbuf2, w_fc3, b_fc3, (float*)d_out, 256, 40);
}
